// MambaBlock_76398878261672
// MI455X (gfx1250) — matrix operand order repaired and checked by reading, not compiled, not run on hardware
//
#include <hip/hip_runtime.h>
#include <hip/hip_bf16.h>

// ---------------------------------------------------------------------------
// Mamba dual-scan block for MI455X (gfx1250), wave32, WMMA f16->f32 GEMMs.
// ---------------------------------------------------------------------------

typedef __attribute__((ext_vector_type(16))) _Float16 v16h;
typedef __attribute__((ext_vector_type(8)))  _Float16 v8h;
typedef __attribute__((ext_vector_type(8)))  float    v8f;

#define DEV __device__ __forceinline__

static constexpr int BB = 4, CC = 64, TT = 400, FF = 201;
static constexpr int DI = 128;           // D_INNER
static constexpr int DS = 16;            // D_STATE
static constexpr int DBL_PAD = 48;       // 36 rounded to 3 WMMA n-tiles

DEV v8f wmma_ff(v16h a, v16h b, v8f c) {
  // D = A(16x32 f16) * B(32x16 f16) + C(16x16 f32)
  return __builtin_amdgcn_wmma_f32_16x16x32_f16(false, a, false, b, (short)0, c,
                                                false, false);
}

// A-fragment (16x32 f16 slab) for this lane from a row-major f16 tile.
DEV v16h afrag(const _Float16* rowbase, int kbase, int hi) {
  const v8h lo = *(const v8h*)(rowbase + kbase + hi * 16);
  const v8h hv = *(const v8h*)(rowbase + kbase + hi * 16 + 8);
  v16h r;
#pragma unroll
  for (int i = 0; i < 8; ++i) { r[i] = lo[i]; r[i + 8] = hv[i]; }
  return r;
}

// B-fragment (32x16) from row-major f16 weight W[out][K]; B[k][n] = W[col0+n][k].
DEV v16h bfrag(const _Float16* w, int cols, int col0, int kbase, int lane) {
  int n = lane & 15, hi = (lane >> 4) & 1;
  return *(const v16h*)(w + (size_t)(col0 + n) * cols + kbase + hi * 16);
}

DEV float siluf(float x) { return x / (1.f + __expf(-x)); }

// ---------------------------------------------------------------------------
// Weight conversion: f32 -> f16 (optionally zero-padded rows)
// ---------------------------------------------------------------------------
__global__ void k_cvt(const float* __restrict__ src, _Float16* __restrict__ dst, int n) {
  int i = blockIdx.x * 256 + threadIdx.x;
  if (i < n) dst[i] = (_Float16)src[i];
}

__global__ void k_cvt_pad(const float* __restrict__ src, _Float16* __restrict__ dst,
                          int rows_src, int rows_dst, int cols) {
  int i = blockIdx.x * 256 + threadIdx.x;
  if (i < rows_dst * cols) {
    int r = i / cols;
    dst[i] = (r < rows_src) ? (_Float16)src[i] : (_Float16)0.f;
  }
}

// ---------------------------------------------------------------------------
// Transpose (B,C,T,F) -> token-major t-layout ((B*F+f)*T + t, C)
// ---------------------------------------------------------------------------
__global__ void k_tr_in(const float* __restrict__ x, float* __restrict__ xt) {
  size_t idx = (size_t)blockIdx.x * 256 + threadIdx.x;
  size_t total = (size_t)BB * FF * TT * CC;
  if (idx >= total) return;
  int c = idx & 63;
  size_t rest = idx >> 6;
  int t = (int)(rest % TT); rest /= TT;
  int f = (int)(rest % FF);
  int b = (int)(rest / FF);
  xt[idx] = x[(((size_t)b * CC + c) * TT + t) * FF + f];
}

// ---------------------------------------------------------------------------
// K1: LayerNorm(64) + in-proj (64 -> 256) via WMMA. One wave per 16-token tile.
// D fragments staged through LDS -> coalesced b128 stores of u0/res (f16).
// ---------------------------------------------------------------------------
__global__ void k_ln_inproj(const float* __restrict__ xin,
                            const float* __restrict__ lnw, const float* __restrict__ lnb,
                            const _Float16* __restrict__ inw16,   // 256x64 f16
                            _Float16* __restrict__ u0, _Float16* __restrict__ res,
                            int L, int TPS) {
  __shared__ _Float16 sm_ln[16 * 72];    // LN'd inputs: 16 tok x 64 ch
  __shared__ _Float16 sm_out[16 * 264];  // staged outputs: 16 tok x 256 (pad 264)
  int tile = blockIdx.x;
  int seq = tile / TPS, l0 = (tile % TPS) * 16;
  int lane = threadIdx.x;
  size_t base = (size_t)seq * L;

  // LayerNorm: lane pair handles one token (32 channels per half-lane).
  int tk = lane >> 1, hh = lane & 1;
  int l = l0 + tk;
  float vals[32];
  float s = 0.f, ss = 0.f;
  if (l < L) {
    const float4* r4 = (const float4*)(xin + (base + l) * 64 + hh * 32);
#pragma unroll
    for (int i = 0; i < 8; ++i) {
      float4 v = r4[i];
      vals[4*i+0] = v.x; vals[4*i+1] = v.y; vals[4*i+2] = v.z; vals[4*i+3] = v.w;
      s  += v.x + v.y + v.z + v.w;
      ss += v.x*v.x + v.y*v.y + v.z*v.z + v.w*v.w;
    }
  } else {
#pragma unroll
    for (int i = 0; i < 32; ++i) vals[i] = 0.f;
  }
  s  += __shfl_xor(s, 1, 32);
  ss += __shfl_xor(ss, 1, 32);
  float mu = s * (1.f / 64.f);
  float var = ss * (1.f / 64.f) - mu * mu;
  float rs = rsqrtf(var + 1e-5f);
  const float4* lw4 = (const float4*)lnw + hh * 8;
  const float4* lb4 = (const float4*)lnb + hh * 8;
#pragma unroll
  for (int i = 0; i < 8; ++i) {
    float4 wv = lw4[i], bv = lb4[i];
    int c = hh * 32 + i * 4;
    sm_ln[tk * 72 + c + 0] = (_Float16)((vals[4*i+0] - mu) * rs * wv.x + bv.x);
    sm_ln[tk * 72 + c + 1] = (_Float16)((vals[4*i+1] - mu) * rs * wv.y + bv.y);
    sm_ln[tk * 72 + c + 2] = (_Float16)((vals[4*i+2] - mu) * rs * wv.z + bv.z);
    sm_ln[tk * 72 + c + 3] = (_Float16)((vals[4*i+3] - mu) * rs * wv.w + bv.w);
  }
  __syncthreads();

  int m = lane & 15, hi = lane >> 4, n = lane & 15;
  v16h a0 = afrag(sm_ln + m * 72, 0, hi);
  v16h a1 = afrag(sm_ln + m * 72, 32, hi);

  for (int nt = 0; nt < 16; ++nt) {
    v8f c = {};
    c = wmma_ff(a0, bfrag(inw16, 64, nt * 16, 0, lane), c);
    c = wmma_ff(a1, bfrag(inw16, 64, nt * 16, 32, lane), c);
#pragma unroll
    for (int i = 0; i < 8; ++i)
      sm_out[(i + hi * 8) * 264 + nt * 16 + n] = (_Float16)c[i];
  }
  __syncthreads();

  // Coalesced writer: 16 tokens x 32 chunks of 8 halves.
  for (int idx = lane; idx < 16 * 32; idx += 32) {
    int rtk = idx >> 5, ck = idx & 31;
    int lr = l0 + rtk;
    if (lr < L) {
      size_t tok = base + lr;
      v8h v = *(const v8h*)(sm_out + rtk * 264 + ck * 8);
      if (ck < 16) *(v8h*)(u0  + tok * DI + ck * 8)        = v;
      else         *(v8h*)(res + tok * DI + (ck - 16) * 8) = v;
    }
  }
}

// ---------------------------------------------------------------------------
// K2: causal depthwise conv(4) + SiLU, then x-proj (128 -> 48pad) via WMMA.
// u0 slab (19 rows incl. halo) staged via b128 loads; conv runs out of LDS
// with per-channel tap weights hoisted to registers.
// ---------------------------------------------------------------------------
__global__ void k_conv_xproj(const _Float16* __restrict__ u0,
                             const float* __restrict__ convw,  // (4,1,128)
                             const float* __restrict__ convb,  // (128)
                             const _Float16* __restrict__ xp16, // 48x128 f16 (padded)
                             float* __restrict__ u, float* __restrict__ dbl,
                             int L, int TPS) {
  __shared__ _Float16 smu[19 * 136];  // raw u0 slab: rows l0-3 .. l0+15
  __shared__ _Float16 sm[16 * 136];   // conv+silu tile (f16)
  __shared__ float    sm_d[16 * 48];  // staged dbl outputs
  int tile = blockIdx.x;
  int seq = tile / TPS, l0 = (tile % TPS) * 16;
  int lane = threadIdx.x;
  size_t base = (size_t)seq * L;

  // Cooperative vector load of the slab (zero-padded out of range).
  for (int idx = lane; idx < 19 * 16; idx += 32) {
    int r = idx >> 4, ck = idx & 15;
    int l = l0 - 3 + r;
    v8h v;
#pragma unroll
    for (int i = 0; i < 8; ++i) v[i] = (_Float16)0.f;
    if (l >= 0 && l < L) v = *(const v8h*)(u0 + (base + l) * DI + ck * 8);
    *(v8h*)(smu + r * 136 + ck * 8) = v;
  }
  __syncthreads();

  // Depthwise conv(4) + SiLU from LDS; each lane owns 4 fixed channels.
#pragma unroll
  for (int q = 0; q < 4; ++q) {
    int ch = lane + q * 32;
    float cw0 = convw[0 * DI + ch], cw1 = convw[1 * DI + ch];
    float cw2 = convw[2 * DI + ch], cw3 = convw[3 * DI + ch];
    float cb  = convb[ch];
    for (int tk = 0; tk < 16; ++tk) {
      int l = l0 + tk;                 // uniform branch across the wave
      float v = 0.f;
      if (l < L) {
        float acc = cb
          + cw0 * (float)smu[(tk + 0) * 136 + ch]
          + cw1 * (float)smu[(tk + 1) * 136 + ch]
          + cw2 * (float)smu[(tk + 2) * 136 + ch]
          + cw3 * (float)smu[(tk + 3) * 136 + ch];
        v = siluf(acc);
        u[(base + l) * DI + ch] = v;
      }
      sm[tk * 136 + ch] = (_Float16)v;
    }
  }
  __syncthreads();

  int m = lane & 15, hi = lane >> 4, n = lane & 15;
  v16h a[4];
#pragma unroll
  for (int q = 0; q < 4; ++q) a[q] = afrag(sm + m * 136, q * 32, hi);

  for (int nt = 0; nt < 3; ++nt) {
    v8f c = {};
#pragma unroll
    for (int q = 0; q < 4; ++q)
      c = wmma_ff(a[q], bfrag(xp16, 128, nt * 16, q * 32, lane), c);
#pragma unroll
    for (int i = 0; i < 8; ++i)
      sm_d[(i + hi * 8) * 48 + nt * 16 + n] = c[i];
  }
  __syncthreads();

  // Coalesced writer: 16 tokens x 12 float4 chunks.
  for (int idx = lane; idx < 16 * 12; idx += 32) {
    int rtk = idx / 12, ck = idx % 12;
    int lr = l0 + rtk;
    if (lr < L)
      *(float4*)(dbl + (base + lr) * DBL_PAD + ck * 4) =
          *(const float4*)(sm_d + rtk * 48 + ck * 4);
  }
}

// ---------------------------------------------------------------------------
// K3: selective scan. One block per sequence, 128 threads (one per inner dim).
// Fuses dt-proj + softplus and gating y * silu(res). Prefetches 8 steps ahead.
// ---------------------------------------------------------------------------
__global__ void k_scan(const float* __restrict__ u, const float* __restrict__ dbl,
                       const _Float16* __restrict__ res,
                       const float* __restrict__ dtw,  // (128,4)
                       const float* __restrict__ dtb,  // (128)
                       const float* __restrict__ Alog, // (128,16)
                       const float* __restrict__ Dp,   // (128)
                       _Float16* __restrict__ g, int L) {
  __shared__ float sm[36];
  int seq = blockIdx.x;
  int d = threadIdx.x;

  float A[DS];
#pragma unroll
  for (int n2 = 0; n2 < DS; ++n2) A[n2] = -__expf(Alog[d * DS + n2]);
  float w0 = dtw[d * 4 + 0], w1 = dtw[d * 4 + 1];
  float w2 = dtw[d * 4 + 2], w3 = dtw[d * 4 + 3];
  float bb = dtb[d], Dd = Dp[d];
  float h[DS];
#pragma unroll
  for (int n2 = 0; n2 < DS; ++n2) h[n2] = 0.f;

  size_t base = (size_t)seq * L;
  for (int l = 0; l < L; ++l) {
    size_t tok = base + l;
    // Prefetch (global_prefetch_b8) a few serial steps ahead to hide latency.
    if (l + 8 < L) {
      size_t tp = tok + 8;
      if (d < 3)         __builtin_prefetch(dbl + tp * DBL_PAD + d * 16, 0, 0);
      if ((d & 31) == 0) __builtin_prefetch(u   + tp * DI + d, 0, 0);
      if ((d & 63) == 0) __builtin_prefetch(res + tp * DI + d, 0, 0);
    }
    if (d < 36) sm[d] = dbl[tok * DBL_PAD + d];
    __syncthreads();
    float x = sm[0] * w0 + sm[1] * w1 + sm[2] * w2 + sm[3] * w3 + bb;
    float delta = (x > 20.f) ? x : logf(1.f + __expf(x));   // softplus
    float uu = u[tok * DI + d];
    float du = delta * uu;
    float y = 0.f;
#pragma unroll
    for (int n2 = 0; n2 < DS; ++n2) {
      float dA = __expf(delta * A[n2]);
      h[n2] = dA * h[n2] + du * sm[4 + n2];
      y += h[n2] * sm[20 + n2];
    }
    y += uu * Dd;
    float r = (float)res[tok * DI + d];
    g[tok * DI + d] = (_Float16)(y * siluf(r));
    __syncthreads();
  }
}

// ---------------------------------------------------------------------------
// K4: out-proj (128 -> 64) via WMMA + residual + layout permute, LDS-staged.
// dir 0: seq=(b*F+f), l=t  -> write f-layout token (b*T+t)*F+f   (row writes)
// dir 1: seq=(b*T+t), l=f  -> write final (B,C,T,F)              (col writes)
// ---------------------------------------------------------------------------
__global__ void k_outproj(const _Float16* __restrict__ g,
                          const _Float16* __restrict__ ow16, // 64x128 f16
                          const float* __restrict__ resid,
                          float* __restrict__ out,
                          int L, int TPS, int dir) {
  __shared__ float sm_o[16 * 64];
  int tile = blockIdx.x;
  int seq = tile / TPS, l0 = (tile % TPS) * 16;
  int lane = threadIdx.x;
  int m = lane & 15, hi = lane >> 4, n = lane & 15;
  size_t base = (size_t)seq * L;

  int lA = l0 + m;
  bool va = (lA < L);
  const _Float16* arow = g + (base + (va ? lA : 0)) * DI;
  v16h a[4];
#pragma unroll
  for (int q = 0; q < 4; ++q) {
    if (va) a[q] = afrag(arow, q * 32, hi);
    else {
      v16h z;
#pragma unroll
      for (int i = 0; i < 16; ++i) z[i] = (_Float16)0.f;
      a[q] = z;
    }
  }

  for (int nt = 0; nt < 4; ++nt) {
    v8f c = {};
#pragma unroll
    for (int q = 0; q < 4; ++q)
      c = wmma_ff(a[q], bfrag(ow16, 128, nt * 16, q * 32, lane), c);
#pragma unroll
    for (int i = 0; i < 8; ++i)
      sm_o[(i + hi * 8) * 64 + nt * 16 + n] = c[i];
  }
  __syncthreads();

  if (dir == 0) {
    int b = seq / FF, f = seq % FF;
    // Row-contiguous destination: 16 tokens x 16 float4 chunks.
    for (int idx = lane; idx < 16 * 16; idx += 32) {
      int rtk = idx >> 4, ck = idx & 15;
      int lr = l0 + rtk;
      if (lr < L) {
        size_t tok = base + lr;
        float4 v = *(const float4*)(sm_o + rtk * 64 + ck * 4);
        float4 r = *(const float4*)(resid + tok * CC + ck * 4);
        v.x += r.x; v.y += r.y; v.z += r.z; v.w += r.w;
        *(float4*)(out + ((size_t)(b * TT + lr) * FF + f) * CC + ck * 4) = v;
      }
    }
  } else {
    int b = seq / TT, t = seq % TT;
    int cnt = L - l0; if (cnt > 16) cnt = 16;
    // Destination contiguous along the sequence axis: write per-column runs.
    for (int col = lane; col < CC; col += 32) {
      float* dst = out + (((size_t)b * CC + col) * TT + t) * FF + l0;
      for (int rtk = 0; rtk < cnt; ++rtk)
        dst[rtk] = sm_o[rtk * 64 + col] + resid[(base + l0 + rtk) * CC + col];
    }
  }
}

// ---------------------------------------------------------------------------
// Host launch
// ---------------------------------------------------------------------------
extern "C" void kernel_launch(void* const* d_in, const int* in_sizes, int n_in,
                              void* d_out, int out_size, void* d_ws, size_t ws_size,
                              hipStream_t stream) {
  (void)in_sizes; (void)n_in; (void)out_size; (void)ws_size;

  const float* x      = (const float*)d_in[0];
  const float* ln_w   = (const float*)d_in[1];
  const float* ln_b   = (const float*)d_in[2];
  const float* inw[2]   = {(const float*)d_in[3],  (const float*)d_in[12]};
  const float* convw[2] = {(const float*)d_in[4],  (const float*)d_in[13]};
  const float* convb[2] = {(const float*)d_in[5],  (const float*)d_in[14]};
  const float* xpw[2]   = {(const float*)d_in[6],  (const float*)d_in[15]};
  const float* dtw[2]   = {(const float*)d_in[7],  (const float*)d_in[16]};
  const float* dtb[2]   = {(const float*)d_in[8],  (const float*)d_in[17]};
  const float* Alog[2]  = {(const float*)d_in[9],  (const float*)d_in[18]};
  const float* Dp[2]    = {(const float*)d_in[10], (const float*)d_in[19]};
  const float* outw[2]  = {(const float*)d_in[11], (const float*)d_in[20]};

  const size_t NT = (size_t)BB * FF * TT;   // 321600 tokens (both directions)

  char* ws = (char*)d_ws;
  size_t off = 0;
  auto carve = [&](size_t bytes) -> char* {
    size_t r = (off + 255) & ~(size_t)255;
    off = r + bytes;
    return ws + r;
  };
  float*     xt   = (float*)    carve(NT * CC * sizeof(float));
  float*     xf   = (float*)    carve(NT * CC * sizeof(float));
  _Float16*  u0g  = (_Float16*) carve(NT * DI * sizeof(_Float16)); // u0, reused as g
  _Float16*  resb = (_Float16*) carve(NT * DI * sizeof(_Float16));
  float*     uu   = (float*)    carve(NT * DI * sizeof(float));
  float*     dbl  = (float*)    carve(NT * DBL_PAD * sizeof(float));
  _Float16*  inw16[2] = {(_Float16*)carve(256 * 64 * 2),  (_Float16*)carve(256 * 64 * 2)};
  _Float16*  xp16[2]  = {(_Float16*)carve(48 * 128 * 2),  (_Float16*)carve(48 * 128 * 2)};
  _Float16*  ow16[2]  = {(_Float16*)carve(64 * 128 * 2),  (_Float16*)carve(64 * 128 * 2)};

  for (int dir = 0; dir < 2; ++dir) {
    k_cvt<<<(256 * 64 + 255) / 256, 256, 0, stream>>>(inw[dir], inw16[dir], 256 * 64);
    k_cvt_pad<<<(48 * 128 + 255) / 256, 256, 0, stream>>>(xpw[dir], xp16[dir], 36, 48, 128);
    k_cvt<<<(64 * 128 + 255) / 256, 256, 0, stream>>>(outw[dir], ow16[dir], 64 * 128);
  }

  {
    size_t total = NT * CC;
    k_tr_in<<<(unsigned)((total + 255) / 256), 256, 0, stream>>>(x, xt);
  }

  for (int dir = 0; dir < 2; ++dir) {
    const int Nseq = (dir == 0) ? BB * FF : BB * TT;   // 804 / 1600
    const int L    = (dir == 0) ? TT : FF;             // 400 / 201
    const int TPS  = (L + 15) / 16;                    // 25 / 13
    const int tiles = Nseq * TPS;
    const float* xin = (dir == 0) ? xt : xf;
    float* xout      = (dir == 0) ? xf : (float*)d_out;

    k_ln_inproj<<<tiles, 32, 0, stream>>>(xin, ln_w, ln_b, inw16[dir],
                                          u0g, resb, L, TPS);
    k_conv_xproj<<<tiles, 32, 0, stream>>>(u0g, convw[dir], convb[dir], xp16[dir],
                                           uu, dbl, L, TPS);
    k_scan<<<Nseq, DI, 0, stream>>>(uu, dbl, resb, dtw[dir], dtb[dir],
                                    Alog[dir], Dp[dir], u0g /* g reuses u0 */, L);
    k_outproj<<<tiles, 32, 0, stream>>>(u0g, ow16[dir], xin, xout, L, TPS, dir);
  }
}
